// LiquidActorCritic_86466281603559
// MI455X (gfx1250) — matrix operand order repaired and checked by reading, not compiled, not run on hardware
//
#include <hip/hip_runtime.h>
#include <hip/hip_bf16.h>

#define B_  2048
#define T_  512
#define D_  256
#define H_  128
#define K2_ 256   // 2H = concat(I, h) width
#define CD_ 6

typedef __bf16 bf16_t;
typedef __bf16 v16bf __attribute__((ext_vector_type(16)));
typedef __bf16 v8bf  __attribute__((ext_vector_type(8)));
typedef float  v8f   __attribute__((ext_vector_type(8)));

// Workspace layout:
//   [0, 5*32768 bf16)           : swizzled bf16 weights (Wp, Wf1, Wf2, Wta, Wtb)
//   [WSW_BYTES, +2048*128 f32)  : x_last
#define WSW_BYTES (5 * 32768 * 2)

// Branch-free activations: v_exp_f32 (exp2) + v_rcp_f32, no EXEC divergence.
// Saturate correctly: x->+inf => 1, x->-inf => -1 (exp2 under/overflow + rcp).
__device__ __forceinline__ float fast_tanh(float x) {
  float e = __builtin_amdgcn_exp2f(x * 2.8853900817779268f);  // e^(2x)
  return 1.0f - 2.0f * __builtin_amdgcn_rcpf(e + 1.0f);
}
__device__ __forceinline__ float fast_sigmoid(float x) {
  float e = __builtin_amdgcn_exp2f(x * -1.4426950408889634f); // e^(-x)
  return __builtin_amdgcn_rcpf(1.0f + e);
}

// ---------------------------------------------------------------------------
// Phase 0: fp32 weights -> bf16, pre-swizzled into the WMMA B-operand layout
// (16x16x32 bf16): lane -> N = ntile*16 + (lane&15),
// element e -> K = kstep*32 + (e&7) + (e>=8 ? 16 : 0) + (lane>=16 ? 8 : 0).
// One fragment = 32 lanes x 16 bf16 contiguous -> each lane loads 32B.
// ---------------------------------------------------------------------------
__global__ __launch_bounds__(256) void swizzle_weights_k(
    const float* __restrict__ Wp,  const float* __restrict__ Wf1,
    const float* __restrict__ Wf2, const float* __restrict__ Wta,
    const float* __restrict__ Wtb, bf16_t* __restrict__ wsw) {
  int idx = blockIdx.x * 256 + threadIdx.x;
  if (idx >= 5 * 32768) return;
  int mat = idx >> 15;
  int rem = idx & 32767;
  int e     = rem & 15;
  int lane  = (rem >> 4) & 31;
  int frag  = rem >> 9;          // kstep*8 + ntile
  int ntile = frag & 7;
  int kstep = frag >> 3;
  int k = kstep * 32 + (e & 7) + ((e & 8) ? 16 : 0) + ((lane & 16) ? 8 : 0);
  int n = ntile * 16 + (lane & 15);
  const float* W = (mat == 0) ? Wp : (mat == 1) ? Wf1 : (mat == 2) ? Wf2
                 : (mat == 3) ? Wta : Wtb;
  wsw[idx] = (bf16_t)W[k * H_ + n];
}

// ---------------------------------------------------------------------------
// Phase 1: fused CfC scan. 128 blocks x 256 threads (8 waves).
// Block = one 16-row batch tile; wave w owns output columns [16w, 16w+16).
// ALL five weight matrices live in VGPRs (40 fragments = 320 VGPRs/wave,
// gfx1250 wave32 supports 1024 VGPRs via MSB addressing). LDS holds only the
// 16x256 x-tile and the 16x256 obs staging tile (16 KB).
// ---------------------------------------------------------------------------
__global__ __launch_bounds__(256) void cfc_scan_k(
    const float* __restrict__ obs, const bf16_t* __restrict__ wsw,
    const float* __restrict__ bp,  const float* __restrict__ bf1,
    const float* __restrict__ bf2, const float* __restrict__ bta,
    const float* __restrict__ btb, float* __restrict__ xlast) {
  __shared__ bf16_t xTile[16 * K2_];    // x = [I, h], row-major
  __shared__ bf16_t obsTile[16 * K2_];  // obs staging (bf16)

  const int tid  = threadIdx.x;
  const int lane = tid & 31;
  const int wv   = tid >> 5;          // wave id == ntile, 0..7
  const int b0   = blockIdx.x * 16;   // batch tile base row

  // Pin all weight fragments for this wave's ntile in registers.
  // wfrag[0] = Wp, wfrag[1..4] = Wf1, Wf2, Wta, Wtb; 8 k-steps each.
  v16bf wfrag[5][8];
#pragma unroll
  for (int m = 0; m < 5; ++m)
#pragma unroll
    for (int ks = 0; ks < 8; ++ks)
      wfrag[m][ks] = *(const v16bf*)(
          wsw + (size_t)m * 32768 + (size_t)((ks * 8 + wv) * 32 + lane) * 16);

  // Zero the h half of the x tile (h0 = 0).
  for (int i = tid; i < 16 * H_; i += 256)
    xTile[(i >> 7) * K2_ + H_ + (i & 127)] = (bf16_t)0.0f;

  // Per-lane fixed output column and its biases (registers).
  const int   cn   = wv * 16 + (lane & 15);
  const float bpc  = bp[cn];
  const float b1c  = bf1[cn];
  const float b2c  = bf2[cn];
  const float btac = bta[cn];
  const float btbc = btb[cn];

  const int m_lo = lane & 15;               // A-operand M row
  const int hi8  = (lane & 16) ? 8 : 0;     // A: +8 K offset; C/D: +8 M offset

  __syncthreads();

  for (int t = 0; t < T_; ++t) {
    // --- stage obs[b0..b0+15, t, :] as bf16 into LDS (row-major) -----------
    {
      const int row = tid >> 4;
      const int cb  = (tid & 15) * 16;
      const float* src = obs + ((size_t)(b0 + row) * T_ + t) * D_ + cb;
      bf16_t tmp[16];
#pragma unroll
      for (int i = 0; i < 16; ++i) tmp[i] = (bf16_t)src[i];
      v8bf* dst = (v8bf*)(obsTile + row * K2_ + cb);
      dst[0] = *(v8bf*)&tmp[0];
      dst[1] = *(v8bf*)&tmp[8];
      if (t + 1 < T_) __builtin_prefetch(src + D_, 0, 0);  // next timestep
    }
    __syncthreads();  // (A) obsTile ready; prev-step h writes visible

    // --- I = tanh(obs_t @ Wp + bp), Wp fragments in registers --------------
    v8f accI = {};
#pragma unroll
    for (int ks = 0; ks < 8; ++ks) {
      const bf16_t* ab = obsTile + m_lo * K2_ + ks * 32 + hi8;
      v8bf g0 = *(const v8bf*)ab;           // K = base+0..7
      v8bf g1 = *(const v8bf*)(ab + 16);    // K = base+16..23
      v16bf a = __builtin_shufflevector(g0, g1, 0,1,2,3,4,5,6,7,
                                        8,9,10,11,12,13,14,15);
      accI = __builtin_amdgcn_wmma_f32_16x16x32_bf16(
                 false, a, false, wfrag[0][ks], (short)0, accI, false, false);
    }
#pragma unroll
    for (int r = 0; r < 8; ++r) {
      float v = fast_tanh(accI[r] + bpc);
      xTile[(r + hi8) * K2_ + cn] = (bf16_t)v;   // I half of x
    }
    __syncthreads();  // (B) full x = [I, h] ready

    // --- gate matmuls: ff1, ff2, ta, tb over x (K = 256), B in registers ---
    v8f a1 = {}, a2 = {}, a3 = {}, a4 = {};
#pragma unroll
    for (int ks = 0; ks < 8; ++ks) {
      const bf16_t* ab = xTile + m_lo * K2_ + ks * 32 + hi8;
      v8bf g0 = *(const v8bf*)ab;
      v8bf g1 = *(const v8bf*)(ab + 16);
      v16bf a = __builtin_shufflevector(g0, g1, 0,1,2,3,4,5,6,7,
                                        8,9,10,11,12,13,14,15);
      a1 = __builtin_amdgcn_wmma_f32_16x16x32_bf16(false, a, false, wfrag[1][ks], (short)0, a1, false, false);
      a2 = __builtin_amdgcn_wmma_f32_16x16x32_bf16(false, a, false, wfrag[2][ks], (short)0, a2, false, false);
      a3 = __builtin_amdgcn_wmma_f32_16x16x32_bf16(false, a, false, wfrag[3][ks], (short)0, a3, false, false);
      a4 = __builtin_amdgcn_wmma_f32_16x16x32_bf16(false, a, false, wfrag[4][ks], (short)0, a4, false, false);
    }
    __syncthreads();  // (C) all xTile reads done before h overwrite (WAR)

    // --- liquid gate + state update (branch-free activations) --------------
#pragma unroll
    for (int r = 0; r < 8; ++r) {
      float f1v = fast_tanh(a1[r] + b1c);
      float f2v = fast_tanh(a2[r] + b2c);
      float s   = (a3[r] + btac) + (a4[r] + btbc);  // t_dt == 1.0
      float g   = fast_sigmoid(s);
      float hn  = f1v * (1.0f - g) + g * f2v;
      xTile[(r + hi8) * K2_ + H_ + cn] = (bf16_t)hn; // h half of x
      if (t == T_ - 1)
        xlast[(size_t)(b0 + r + hi8) * H_ + cn] = hn;
    }
  }
}

// ---------------------------------------------------------------------------
// Phase 2: tiny heads (31 output columns, K=128) — bandwidth-trivial VALU.
// Block = one batch row, 32 threads (one column each).
// ---------------------------------------------------------------------------
__global__ __launch_bounds__(32) void heads_k(
    const float* __restrict__ xlast,
    const float* __restrict__ Wmu, const float* __restrict__ bmu,
    const float* __restrict__ log_std,
    const float* __restrict__ Wh0, const float* __restrict__ bh0,
    const float* __restrict__ Wh1, const float* __restrict__ bh1,
    const float* __restrict__ Wh2, const float* __restrict__ bh2,
    const float* __restrict__ Wv,  const float* __restrict__ bv,
    float* __restrict__ out) {
  __shared__ float xr[H_];
  const int b = blockIdx.x, j = threadIdx.x;
  for (int i = j; i < H_; i += 32) xr[i] = xlast[(size_t)b * H_ + i];
  __syncthreads();
  if (j < 6) {                                   // mu
    float s = bmu[j];
    for (int k = 0; k < H_; ++k) s += xr[k] * Wmu[k * 6 + j];
    out[b * 6 + j] = fast_tanh(s);
  } else if (j < 18) {                           // logits0 [B,12]
    int c = j - 6; float s = bh0[c];
    for (int k = 0; k < H_; ++k) s += xr[k] * Wh0[k * 12 + c];
    out[12294 + b * 12 + c] = s;
  } else if (j < 26) {                           // logits1 [B,8]
    int c = j - 18; float s = bh1[c];
    for (int k = 0; k < H_; ++k) s += xr[k] * Wh1[k * 8 + c];
    out[36870 + b * 8 + c] = s;
  } else if (j < 30) {                           // logits2 [B,4]
    int c = j - 26; float s = bh2[c];
    for (int k = 0; k < H_; ++k) s += xr[k] * Wh2[k * 4 + c];
    out[53254 + b * 4 + c] = s;
  } else if (j == 30) {                          // value [B]
    float s = bv[0];
    for (int k = 0; k < H_; ++k) s += xr[k] * Wv[k];
    out[61446 + b] = s;
  } else {                                       // std [6]
    if (b < 6) out[12288 + b] = fmaxf(__expf(log_std[b]), 1e-6f);
  }
}

extern "C" void kernel_launch(void* const* d_in, const int* in_sizes, int n_in,
                              void* d_out, int out_size, void* d_ws, size_t ws_size,
                              hipStream_t stream) {
  (void)in_sizes; (void)n_in; (void)out_size; (void)ws_size;
  const float* obs     = (const float*)d_in[0];
  const float* Wp      = (const float*)d_in[1];
  const float* bp      = (const float*)d_in[2];
  const float* Wf1     = (const float*)d_in[3];
  const float* bf1     = (const float*)d_in[4];
  const float* Wf2     = (const float*)d_in[5];
  const float* bf2     = (const float*)d_in[6];
  const float* Wta     = (const float*)d_in[7];
  const float* bta     = (const float*)d_in[8];
  const float* Wtb     = (const float*)d_in[9];
  const float* btb     = (const float*)d_in[10];
  const float* Wmu     = (const float*)d_in[11];
  const float* bmu     = (const float*)d_in[12];
  const float* log_std = (const float*)d_in[13];
  const float* Wh0     = (const float*)d_in[14];
  const float* bh0     = (const float*)d_in[15];
  const float* Wh1     = (const float*)d_in[16];
  const float* bh1     = (const float*)d_in[17];
  const float* Wh2     = (const float*)d_in[18];
  const float* bh2     = (const float*)d_in[19];
  const float* Wv      = (const float*)d_in[20];
  const float* bv      = (const float*)d_in[21];
  float* out = (float*)d_out;

  bf16_t* wsw   = (bf16_t*)d_ws;
  float*  xlast = (float*)((char*)d_ws + WSW_BYTES);

  swizzle_weights_k<<<(5 * 32768 + 255) / 256, 256, 0, stream>>>(
      Wp, Wf1, Wf2, Wta, Wtb, wsw);

  cfc_scan_k<<<B_ / 16, 256, 0, stream>>>(
      obs, wsw, bp, bf1, bf2, bta, btb, xlast);

  heads_k<<<B_, 32, 0, stream>>>(
      xlast, Wmu, bmu, log_std, Wh0, bh0, Wh1, bh1, Wh2, bh2, Wv, bv, out);
}
